// NeuralLongTermMemory_27135603376244
// MI455X (gfx1250) — matrix operand order repaired and checked by reading, not compiled, not run on hardware
//
#include <hip/hip_runtime.h>
#include <hip/hip_bf16.h>
#include <math.h>

typedef __attribute__((ext_vector_type(16))) __bf16 v16bf;
typedef __attribute__((ext_vector_type(8)))  float  v8f;

#define Ss 1024
#define Hh 1024
#define Dd 256
#define Tt 512

// ---------- helpers ----------
__device__ __forceinline__ unsigned short f2bf(float f) {
    union { float f; unsigned u; } x; x.f = f;
    unsigned r = x.u + 0x7FFFu + ((x.u >> 16) & 1u);
    return (unsigned short)(r >> 16);
}
__device__ __forceinline__ float sigmoidf_(float x) { return 1.0f / (1.0f + __expf(-x)); }

__device__ __forceinline__ float block_reduce(float v, float* sbuf) {
    int tid = threadIdx.x;
    sbuf[tid] = v;
    __syncthreads();
    for (int s = blockDim.x >> 1; s > 0; s >>= 1) {
        if (tid < s) sbuf[tid] += sbuf[tid + s];
        __syncthreads();
    }
    float r = sbuf[0];
    __syncthreads();
    return r;
}

// ---------- fp32 -> bf16 convert ----------
__global__ void cvt_bf16_k(const float* __restrict__ in, unsigned short* __restrict__ out, int n) {
    int i = blockIdx.x * blockDim.x + threadIdx.x;
    if (i < n) out[i] = f2bf(in[i]);
}

// ---------- bf16 NT GEMM: C[M,N] = sum_k A[m,k]*B[n,k]; A:(MxK) B:(NxK) row-major ----------
// 256 threads = 8 waves (wave32), block tile 256x128, wave tile 64x64 (4x4 WMMA frags).
// Double-buffered LDS staging via gfx1250 async copy (global_load_async_to_lds_b128,
// tracked by ASYNCcnt) -- no VGPR round trip, compute of tile i overlaps fetch of i+1.
union Frag16 { v16bf v; uint4 q[2]; };

__global__ __launch_bounds__(256) void gemm_nt_bf16(
    const unsigned short* __restrict__ A, const unsigned short* __restrict__ B,
    float* __restrict__ C, int M, int N, int K) {
    __shared__ unsigned short sA[2][256 * 32];
    __shared__ unsigned short sB[2][128 * 32];
    const int tid  = threadIdx.x;
    const int lane = tid & 31, wave = tid >> 5;
    const int wm = wave >> 1, wn = wave & 1;          // 4x2 wave grid
    const int bM = blockIdx.y * 256, bN = blockIdx.x * 128;

    // staging slices: thread t owns A row (bM+t), 32 halves/iter; B row (bN+t/2), 16 halves
    const unsigned short* gA = A + (size_t)(bM + tid) * K;
    const unsigned short* gB = B + (size_t)(bN + (tid >> 1)) * K + (tid & 1) * 16;

    unsigned aoff[2], boff[2];
    aoff[0] = (unsigned)(uintptr_t)&sA[0][tid * 32];
    aoff[1] = (unsigned)(uintptr_t)&sA[1][tid * 32];
    boff[0] = (unsigned)(uintptr_t)&sB[0][(tid >> 1) * 32 + (tid & 1) * 16];
    boff[1] = (unsigned)(uintptr_t)&sB[1][(tid >> 1) * 32 + (tid & 1) * 16];

    auto stage = [&](int buf, int k0) {
        const unsigned short* a0 = gA + k0;
        const unsigned short* b0 = gB + k0;
        asm volatile("global_load_async_to_lds_b128 %0, %1, off" :: "v"(aoff[buf]),       "v"(a0)      : "memory");
        asm volatile("global_load_async_to_lds_b128 %0, %1, off" :: "v"(aoff[buf] + 16u), "v"(a0 + 8)  : "memory");
        asm volatile("global_load_async_to_lds_b128 %0, %1, off" :: "v"(aoff[buf] + 32u), "v"(a0 + 16) : "memory");
        asm volatile("global_load_async_to_lds_b128 %0, %1, off" :: "v"(aoff[buf] + 48u), "v"(a0 + 24) : "memory");
        asm volatile("global_load_async_to_lds_b128 %0, %1, off" :: "v"(boff[buf]),       "v"(b0)      : "memory");
        asm volatile("global_load_async_to_lds_b128 %0, %1, off" :: "v"(boff[buf] + 16u), "v"(b0 + 8)  : "memory");
    };

    // prologue: fill buffer 0
    stage(0, 0);
    asm volatile("s_wait_asynccnt 0" ::: "memory");
    __syncthreads();

    v8f acc[4][4] = {};
    const int mla = lane & 15;
    const int ka  = (lane >> 4) * 8;   // A frag: lanes<16 K=[0..7]+[16..23]; lanes>=16 K=[8..15]+[24..31]
    const int kb  = (lane >> 4) * 8;   // B frag: lanes<16 K=[0..7]+[16..23]; lanes>=16 K=[8..15]+[24..31]
    const int nk  = K >> 5;

    for (int it = 0; it < nk; it++) {
        const int cur = it & 1;
        if (it + 1 < nk) stage(cur ^ 1, (it + 1) * 32);

        Frag16 a[4];
        #pragma unroll
        for (int i = 0; i < 4; i++) {
            int row = wm * 64 + i * 16 + mla;
            a[i].q[0] = *(const uint4*)&sA[cur][row * 32 + ka];
            a[i].q[1] = *(const uint4*)&sA[cur][row * 32 + ka + 16];
        }
        #pragma unroll
        for (int j = 0; j < 4; j++) {
            Frag16 b;
            int col = wn * 64 + j * 16 + mla;
            b.q[0] = *(const uint4*)&sB[cur][col * 32 + kb];
            b.q[1] = *(const uint4*)&sB[cur][col * 32 + kb + 16];
            #pragma unroll
            for (int i = 0; i < 4; i++)
                acc[i][j] = __builtin_amdgcn_wmma_f32_16x16x32_bf16(
                    false, a[i].v, false, b.v, (short)0, acc[i][j], false, false);
        }
        // next buffer filled + everyone done reading current before it is reused
        asm volatile("s_wait_asynccnt 0" ::: "memory");
        __syncthreads();
    }

    // C layout: vgpr r -> row = r + (lane>=16 ? 8 : 0), col = lane&15
    const int rb = (lane >> 4) * 8;
    #pragma unroll
    for (int i = 0; i < 4; i++) {
        #pragma unroll
        for (int j = 0; j < 4; j++) {
            int col = bN + wn * 64 + j * 16 + mla;
            int rowbase = bM + wm * 64 + i * 16 + rb;
            #pragma unroll
            for (int r = 0; r < 8; r++)
                C[(size_t)(rowbase + r) * N + col] = acc[i][j][r];
        }
    }
}

// ---------- depthwise causal conv (k=4, left pad 3) ----------
__global__ void dwconv_k(const float* __restrict__ x, const float* __restrict__ w,
                         const float* __restrict__ bia, float* __restrict__ y,
                         unsigned short* __restrict__ ybf, int S, int D) {
    int i = blockIdx.x * blockDim.x + threadIdx.x;
    if (i >= S * D) return;
    int s = i / D, d = i % D;
    float acc = bia[d];
    #pragma unroll
    for (int j = 0; j < 4; j++) {
        int t = s + j - 3;
        if (t >= 0) acc += w[d * 4 + j] * x[t * D + d];
    }
    y[i] = acc;
    if (ybf) ybf[i] = f2bf(acc);
}

// ---------- per-token gate scalars ----------
__global__ __launch_bounds__(256) void gates_k(
    const float* __restrict__ Ag, const float* __restrict__ Eg, const float* __restrict__ Tg,
    const float* __restrict__ ab, const float* __restrict__ eb, const float* __restrict__ tb,
    float* __restrict__ alpha, float* __restrict__ eta, float* __restrict__ theta) {
    __shared__ float sbuf[256];
    int s = blockIdx.x, d = threadIdx.x;
    float va = sigmoidf_(Ag[s * Dd + d] + ab[d]);
    float ve = sigmoidf_(Eg[s * Dd + d] + eb[d]);
    float xt = Tg[s * Dd + d] + tb[d];
    float vt = (xt > 20.f) ? xt : log1pf(__expf(xt));
    float ra = block_reduce(va, sbuf);
    float re = block_reduce(ve, sbuf);
    float rt = block_reduce(vt, sbuf);
    if (d == 0) {
        alpha[s] = 0.1f * (ra / (float)Dd);
        eta[s]   = 0.9f * (re / (float)Dd) + 0.05f;
        theta[s] = 0.1f * (rt / (float)Dd);
    }
}

// ---------- elementwise pieces ----------
__global__ void silu_k(const float* __restrict__ Z, float* __restrict__ m,
                       unsigned short* __restrict__ mbf, int n) {
    int i = blockIdx.x * blockDim.x + threadIdx.x;
    if (i >= n) return;
    float z = Z[i], v = z * sigmoidf_(z);
    m[i] = v; mbf[i] = f2bf(v);
}
__global__ void dpred_k(const float* __restrict__ P, const float* __restrict__ v,
                        float* __restrict__ dp, unsigned short* __restrict__ dpbf, int n) {
    int i = blockIdx.x * blockDim.x + threadIdx.x;
    if (i >= n) return;
    float t = 2.f * (P[i] - v[i]);
    dp[i] = t; dpbf[i] = f2bf(t);
}
__global__ void dz_k(const float* __restrict__ DH, const float* __restrict__ Z,
                     float* __restrict__ dz, int n) {
    int i = blockIdx.x * blockDim.x + threadIdx.x;
    if (i >= n) return;
    float z = Z[i], sg = sigmoidf_(z);
    dz[i] = DH[i] * (sg * (1.f + z * (1.f - sg)));
}
__global__ void mul_bf16_k(const float* __restrict__ A, const float* __restrict__ B,
                           unsigned short* __restrict__ out, int n) {
    int i = blockIdx.x * blockDim.x + threadIdx.x;
    if (i < n) out[i] = f2bf(A[i] * B[i]);
}
__global__ void transpose_scale_bf16_k(const float* __restrict__ in, unsigned short* __restrict__ out,
                                       const float* __restrict__ scale, int R, int C) {
    int i = blockIdx.x * blockDim.x + threadIdx.x;
    if (i >= R * C) return;
    int r = i / C, c = i % C;
    float v = in[i];
    if (scale) v *= scale[r];
    out[(size_t)c * R + r] = f2bf(v);
}

// ---------- per-token grad-clip gamma ----------
__global__ __launch_bounds__(256) void norms_k(
    const float* __restrict__ dz, const float* __restrict__ kf,
    const float* __restrict__ dpf, const float* __restrict__ mf, float* __restrict__ gamma) {
    __shared__ float sbuf[256];
    int t = blockIdx.x, d = threadIdx.x;
    float a = dz[t * Dd + d], b = kf[t * Dd + d], c = dpf[t * Dd + d], e = mf[t * Dd + d];
    float s1 = block_reduce(a * a, sbuf);
    float s2 = block_reduce(b * b, sbuf);
    float s3 = block_reduce(c * c, sbuf);
    float s4 = block_reduce(e * e, sbuf);
    if (d == 0) {
        float gn = 0.1f * sqrtf(s1 * s2 + s3 * s4);   // GRAD_SCALE * ||g||
        gn = fmaxf(gn, 1e-6f);
        float clip = fminf(1.0f / gn, 1.0f);          // MAX_GRAD_NORM = 1
        gamma[t] = 0.1f * clip;                       // effective rank-1 coefficient
    }
}

// ---------- scan coefficients: W_t = A_t W0 + B_t S0 + sum_j Cg[t,j] * outer-grads ----------
__global__ __launch_bounds__(512) void coeff_k(
    const float* __restrict__ alpha, const float* __restrict__ eta, const float* __restrict__ theta,
    const float* __restrict__ gamma, float* __restrict__ Cg,
    float* __restrict__ Avec, float* __restrict__ Bvec,
    float* __restrict__ wW, float* __restrict__ wS, float* __restrict__ scal, int c0) {
    int j = threadIdx.x;
    float thj = theta[c0 + j], gj = gamma[j];
    for (int t = 0; t < j; t++) Cg[t * Tt + j] = 0.f;
    float s = 0.f, c = 0.f;
    for (int t = j; t < Tt; t++) {
        float e = eta[c0 + t], a = 1.f - alpha[c0 + t];
        s = (t == j) ? -thj : e * s;   // coeff of g_j inside S_t
        c = a * c + s;                 // coeff of g_j inside W_t
        Cg[t * Tt + j] = c * gj;
    }
    wS[j] = s * gj;   // end-of-chunk S-state weights
    wW[j] = c * gj;   // end-of-chunk W-state weights
    if (j == 0) {
        float A = 1.f, Bc = 0.f, sb = 1.f;
        for (int t = 0; t < Tt; t++) {
            float e = eta[c0 + t], a = 1.f - alpha[c0 + t];
            sb *= e;            // coeff of S0 in S_t
            Bc = a * Bc + sb;   // coeff of S0 in W_t
            A *= a;             // coeff of W0 in W_t
            Avec[t] = A; Bvec[t] = Bc;
        }
        scal[0] = A; scal[1] = Bc; scal[2] = sb;
    }
}

// ---------- state combine ----------
__global__ void state_combine_k(float* __restrict__ W, float* __restrict__ Sst,
                                const float* __restrict__ GW, const float* __restrict__ GS,
                                const float* __restrict__ scal, int n) {
    int i = blockIdx.x * blockDim.x + threadIdx.x;
    if (i >= n) return;
    float Aend = scal[0], Bend = scal[1], Eend = scal[2];
    float w = W[i], s = Sst[i];
    W[i]   = Aend * w + Bend * s + GW[i];
    Sst[i] = Eend * s + GS[i];
}

__global__ void init_state_k(const float* __restrict__ mW1, const float* __restrict__ mW2,
                             float* __restrict__ W1, float* __restrict__ W2,
                             float* __restrict__ S1, float* __restrict__ S2, int n) {
    int i = blockIdx.x * blockDim.x + threadIdx.x;
    if (i >= n) return;
    W1[i] = mW1[i]; W2[i] = mW2[i]; S1[i] = 0.f; S2[i] = 0.f;
}

// ---------- retrieval combines ----------
__global__ void hcombine_k(const float* __restrict__ QW, const float* __restrict__ QS,
                           const float* __restrict__ U, const float* __restrict__ Avec,
                           const float* __restrict__ Bvec, unsigned short* __restrict__ hbf, int n) {
    int i = blockIdx.x * blockDim.x + threadIdx.x;
    if (i >= n) return;
    int t = i / Dd;
    float z = Avec[t] * QW[i] + Bvec[t] * QS[i] + U[i];
    hbf[i] = f2bf(z * sigmoidf_(z));
}
__global__ void rcombine_k(const float* __restrict__ HW, const float* __restrict__ HS,
                           const float* __restrict__ U, const float* __restrict__ Avec,
                           const float* __restrict__ Bvec, unsigned short* __restrict__ rbf, int n) {
    int i = blockIdx.x * blockDim.x + threadIdx.x;
    if (i >= n) return;
    int t = i / Dd;
    rbf[i] = f2bf(Avec[t] * HW[i] + Bvec[t] * HS[i] + U[i]);
}

// ---------- RMSNorm * gate epilogue ----------
__global__ __launch_bounds__(256) void epilogue_k(
    const float* __restrict__ OUT0, const float* __restrict__ GateLin,
    const float* __restrict__ norm_g, const float* __restrict__ gb, float* __restrict__ out) {
    __shared__ float sbuf[256];
    int s = blockIdx.x, tid = threadIdx.x;
    float vals[4], part = 0.f;
    #pragma unroll
    for (int r = 0; r < 4; r++) {
        int h = tid + r * 256;
        float v = OUT0[(size_t)s * Hh + h];
        vals[r] = v; part += v * v;
    }
    float ssum = block_reduce(part, sbuf);
    float rms = rsqrtf(ssum / (float)Hh + 1e-6f);
    #pragma unroll
    for (int r = 0; r < 4; r++) {
        int h = tid + r * 256;
        float g = GateLin[(size_t)s * Hh + h] + gb[h];
        out[(size_t)s * Hh + h] = vals[r] * rms * norm_g[h] * g;
    }
}

// =======================================================================
extern "C" void kernel_launch(void* const* d_in, const int* in_sizes, int n_in,
                              void* d_out, int out_size, void* d_ws, size_t ws_size,
                              hipStream_t stream) {
    (void)in_sizes; (void)n_in; (void)out_size; (void)ws_size;
    const float* hidden = (const float*)d_in[0];
    const float* Wk  = (const float*)d_in[1];
    const float* Wv  = (const float*)d_in[2];
    const float* Wq  = (const float*)d_in[3];
    const float* Wo  = (const float*)d_in[4];
    const float* ckw = (const float*)d_in[5];  const float* ckb = (const float*)d_in[6];
    const float* cvw = (const float*)d_in[7];  const float* cvb = (const float*)d_in[8];
    const float* cqw = (const float*)d_in[9];  const float* cqb = (const float*)d_in[10];
    const float* mW1 = (const float*)d_in[11]; const float* mW2 = (const float*)d_in[12];
    const float* aW  = (const float*)d_in[13]; const float* ab  = (const float*)d_in[14];
    const float* eW  = (const float*)d_in[15]; const float* eb  = (const float*)d_in[16];
    const float* tW  = (const float*)d_in[17]; const float* tb  = (const float*)d_in[18];
    const float* norm_g = (const float*)d_in[19];
    const float* gW  = (const float*)d_in[20]; const float* gb  = (const float*)d_in[21];
    float* out = (float*)d_out;

    const int S = Ss, H = Hh, D = Dd, T = Tt;

    char* base = (char*)d_ws; size_t off = 0;
    auto alloc = [&](size_t bytes) -> void* {
        off = (off + 255) & ~(size_t)255;
        void* p = base + off; off += bytes; return p;
    };
    auto fbuf = [&](size_t n) { return (float*)alloc(n * 4); };
    auto bbuf = [&](size_t n) { return (unsigned short*)alloc(n * 2); };

    // bf16 buffers
    unsigned short* hid_bf = bbuf((size_t)S * H);
    unsigned short* Wk_bf = bbuf((size_t)D * H); unsigned short* Wv_bf = bbuf((size_t)D * H);
    unsigned short* Wq_bf = bbuf((size_t)D * H);
    unsigned short* aW_bf = bbuf((size_t)D * H); unsigned short* eW_bf = bbuf((size_t)D * H);
    unsigned short* tW_bf = bbuf((size_t)D * H);
    unsigned short* gW_bf = bbuf((size_t)H * H);
    unsigned short* Wo_bf = bbuf((size_t)H * D);
    unsigned short* k_bf  = bbuf((size_t)S * D); unsigned short* q_bf = bbuf((size_t)S * D);
    unsigned short* W1_bf = bbuf((size_t)D * D); unsigned short* W2_bf = bbuf((size_t)D * D);
    unsigned short* W2T_bf= bbuf((size_t)D * D);
    unsigned short* S1_bf = bbuf((size_t)D * D); unsigned short* S2_bf = bbuf((size_t)D * D);
    unsigned short* m_bf  = bbuf((size_t)T * D); unsigned short* dp_bf = bbuf((size_t)T * D);
    unsigned short* h_bf  = bbuf((size_t)T * D);
    unsigned short* CS_bf = bbuf((size_t)T * T);
    unsigned short* KT    = bbuf((size_t)D * T); unsigned short* MT  = bbuf((size_t)D * T);
    unsigned short* DZT   = bbuf((size_t)D * T); unsigned short* DPT = bbuf((size_t)D * T);
    unsigned short* DZw   = bbuf((size_t)D * T); unsigned short* DPw = bbuf((size_t)D * T);
    unsigned short* retr_bf = bbuf((size_t)S * D);

    // fp32 buffers
    float* K0 = fbuf((size_t)S * D); float* V0 = fbuf((size_t)S * D); float* Q0 = fbuf((size_t)S * D);
    float* Ag = fbuf((size_t)S * D); float* Eg = fbuf((size_t)S * D); float* Tg = fbuf((size_t)S * D);
    float* GateLin = fbuf((size_t)S * H);
    float* kf = fbuf((size_t)S * D); float* vf = fbuf((size_t)S * D); float* qf = fbuf((size_t)S * D);
    float* alpha = fbuf(S); float* eta = fbuf(S); float* theta = fbuf(S);
    float* W1 = fbuf((size_t)D * D); float* W2 = fbuf((size_t)D * D);
    float* S1 = fbuf((size_t)D * D); float* S2 = fbuf((size_t)D * D);
    float* Z   = fbuf((size_t)T * D); float* mfb = fbuf((size_t)T * D);
    float* Pp  = fbuf((size_t)T * D); float* dpf = fbuf((size_t)T * D);
    float* DHb = fbuf((size_t)T * D); float* dzf = fbuf((size_t)T * D);
    float* gamma = fbuf(T); float* Avec = fbuf(T); float* Bvec = fbuf(T);
    float* wW = fbuf(T); float* wS = fbuf(T); float* scal = fbuf(8);
    float* Cg  = fbuf((size_t)T * T);
    float* Att = fbuf((size_t)T * T);
    float* GW1 = fbuf((size_t)D * D); float* GW2 = fbuf((size_t)D * D);
    float* GS1 = fbuf((size_t)D * D); float* GS2 = fbuf((size_t)D * D);
    float* QW1 = fbuf((size_t)T * D); float* QS1 = fbuf((size_t)T * D); float* U1 = fbuf((size_t)T * D);
    float* HW2 = fbuf((size_t)T * D); float* HS2 = fbuf((size_t)T * D); float* U2 = fbuf((size_t)T * D);
    float* OUT0 = fbuf((size_t)S * H);

    auto cvt = [&](const float* in, unsigned short* o, int n) {
        cvt_bf16_k<<<n / 256, 256, 0, stream>>>(in, o, n);
    };
    auto gemm = [&](const unsigned short* A, const unsigned short* B, float* C, int M, int N, int K) {
        gemm_nt_bf16<<<dim3(N / 128, M / 256), 256, 0, stream>>>(A, B, C, M, N, K);
    };
    auto transp = [&](const float* in, unsigned short* o, const float* sc, int R, int C) {
        transpose_scale_bf16_k<<<(R * C) / 256, 256, 0, stream>>>(in, o, sc, R, C);
    };

    // ---- Phase A: projections, gate nets, gate GEMM ----
    cvt(hidden, hid_bf, S * H);
    cvt(Wk, Wk_bf, D * H); cvt(Wv, Wv_bf, D * H); cvt(Wq, Wq_bf, D * H);
    cvt(aW, aW_bf, D * H); cvt(eW, eW_bf, D * H); cvt(tW, tW_bf, D * H);
    cvt(gW, gW_bf, H * H); cvt(Wo, Wo_bf, H * D);

    gemm(hid_bf, Wk_bf, K0, S, D, H);
    gemm(hid_bf, Wv_bf, V0, S, D, H);
    gemm(hid_bf, Wq_bf, Q0, S, D, H);
    gemm(hid_bf, aW_bf, Ag, S, D, H);
    gemm(hid_bf, eW_bf, Eg, S, D, H);
    gemm(hid_bf, tW_bf, Tg, S, D, H);
    gemm(hid_bf, gW_bf, GateLin, S, H, H);

    dwconv_k<<<(S * D) / 256, 256, 0, stream>>>(K0, ckw, ckb, kf, k_bf, S, D);
    dwconv_k<<<(S * D) / 256, 256, 0, stream>>>(V0, cvw, cvb, vf, (unsigned short*)nullptr, S, D);
    dwconv_k<<<(S * D) / 256, 256, 0, stream>>>(Q0, cqw, cqb, qf, q_bf, S, D);

    gates_k<<<S, 256, 0, stream>>>(Ag, Eg, Tg, ab, eb, tb, alpha, eta, theta);
    init_state_k<<<(D * D) / 256, 256, 0, stream>>>(mW1, mW2, W1, W2, S1, S2, D * D);

    // ---- Phase B: two chunks ----
    for (int c = 0; c < 2; c++) {
        int c0 = c * T;
        const unsigned short* kcb = k_bf + (size_t)c0 * D;
        const unsigned short* qcb = q_bf + (size_t)c0 * D;
        const float* kcf = kf + (size_t)c0 * D;
        const float* vcf = vf + (size_t)c0 * D;

        // chunk-start state snapshots in bf16
        cvt(W1, W1_bf, D * D); cvt(W2, W2_bf, D * D);
        cvt(S1, S1_bf, D * D); cvt(S2, S2_bf, D * D);
        transp(W2, W2T_bf, nullptr, D, D);

        // memory forward + backward at chunk-start weights (rank-1 grads)
        gemm(kcb, W1_bf, Z, T, D, D);
        silu_k<<<(T * D) / 256, 256, 0, stream>>>(Z, mfb, m_bf, T * D);
        gemm(m_bf, W2_bf, Pp, T, D, D);
        dpred_k<<<(T * D) / 256, 256, 0, stream>>>(Pp, vcf, dpf, dp_bf, T * D);
        gemm(dp_bf, W2T_bf, DHb, T, D, D);
        dz_k<<<(T * D) / 256, 256, 0, stream>>>(DHb, Z, dzf, T * D);
        norms_k<<<T, 256, 0, stream>>>(dzf, kcf, dpf, mfb, gamma);
        coeff_k<<<1, 512, 0, stream>>>(alpha, eta, theta, gamma, Cg, Avec, Bvec, wW, wS, scal, c0);

        // transposed bf16 operands
        transp(kcf, KT, nullptr, T, D);
        transp(mfb, MT, nullptr, T, D);
        transp(dzf, DZT, nullptr, T, D);
        transp(dpf, DPT, nullptr, T, D);

        // retrieval layer 1: silu(A_t*qW1 + B_t*qS1 + (Cg o q k^T) @ dz)
        gemm(qcb, W1_bf, QW1, T, D, D);
        gemm(qcb, S1_bf, QS1, T, D, D);
        gemm(qcb, kcb, Att, T, T, D);
        mul_bf16_k<<<(T * T) / 256, 256, 0, stream>>>(Cg, Att, CS_bf, T * T);
        gemm(CS_bf, DZT, U1, T, D, T);
        hcombine_k<<<(T * D) / 256, 256, 0, stream>>>(QW1, QS1, U1, Avec, Bvec, h_bf, T * D);

        // retrieval layer 2: A_t*hW2 + B_t*hS2 + (Cg o h m^T) @ dp
        gemm(h_bf, m_bf, Att, T, T, D);
        mul_bf16_k<<<(T * T) / 256, 256, 0, stream>>>(Cg, Att, CS_bf, T * T);
        gemm(CS_bf, DPT, U2, T, D, T);
        gemm(h_bf, W2_bf, HW2, T, D, D);
        gemm(h_bf, S2_bf, HS2, T, D, D);
        rcombine_k<<<(T * D) / 256, 256, 0, stream>>>(HW2, HS2, U2, Avec, Bvec,
                                                     retr_bf + (size_t)c0 * D, T * D);

        // end-of-chunk state update: sum_j w_j * outer(dz_j,k_j) etc. as DxT @ TxD GEMMs
        transp(dzf, DZw, wW, T, D);
        transp(dpf, DPw, wW, T, D);
        gemm(DZw, KT, GW1, D, D, T);
        gemm(DPw, MT, GW2, D, D, T);
        transp(dzf, DZw, wS, T, D);
        transp(dpf, DPw, wS, T, D);
        gemm(DZw, KT, GS1, D, D, T);
        gemm(DPw, MT, GS2, D, D, T);
        state_combine_k<<<(D * D) / 256, 256, 0, stream>>>(W1, S1, GW1, GS1, scal, D * D);
        state_combine_k<<<(D * D) / 256, 256, 0, stream>>>(W2, S2, GW2, GS2, scal, D * D);
    }

    // ---- output projection + RMSNorm + gate ----
    gemm(retr_bf, Wo_bf, OUT0, S, H, D);
    epilogue_k<<<S, 256, 0, stream>>>(OUT0, GateLin, norm_g, gb, out);
}
